// GDAGDL_model_83150566851176
// MI455X (gfx1250) — matrix operand order repaired and checked by reading, not compiled, not run on hardware
//
#include <hip/hip_runtime.h>
#include <math.h>

typedef __attribute__((ext_vector_type(16))) _Float16 v16h;
typedef __attribute__((ext_vector_type(8)))  float    v8f;

#define NUM_PATCH 32
#define PATCH     2048
#define NPERSEG   64
#define NFREQ     33          // NPERSEG/2+1
#define NFRAME    33          // 1 + PATCH/NPERSEG
#define NPATCHES  1024        // 32 batches * 32 patches
#define LRELU_A   0.1f

// ---------------- workspace layout (units of float = 4 bytes) ----------------
static const size_t OFF_MAG  = 0;        // f32  1024*33*33            = 1,115,136
static const size_t OFF_HI   = 1115136;  // f32  1024*33               =    33,792
static const size_t OFF_H1   = 1148928;  // f32  1024*33*64            = 2,162,688
static const size_t OFF_H2   = 3311616;  // f16  1024*2112 halves      = 1,081,344 f-slots (Y)
static const size_t OFF_W0T  = 4392960;  // f16  1024*2112 halves      = 1,081,344
static const size_t OFF_W1T  = 5474304;  // f16  512*1024              =   262,144
static const size_t OFF_W2T  = 5736448;  // f16  256*512               =    65,536
static const size_t OFF_W3T  = 5801984;  // f16  128*256               =    16,384
static const size_t OFF_WIH  = 5818368;  // f16  1024*128              =    65,536
static const size_t OFF_WHH  = 5883904;  // f16  1024*256              =   131,072
static const size_t OFF_H0   = 6014976;  // f16  1024*1024             =   524,288
static const size_t OFF_HE1  = 6539264;  // f16  1024*512              =   262,144
static const size_t OFF_HE2  = 6801408;  // f16  1024*256              =   131,072
static const size_t OFF_HE3  = 6932480;  // f16  1024*128              =    65,536
static const size_t OFF_G    = 6998016;  // f32  32*1024               =    32,768
static const size_t OFF_LHH  = 7030784;  // f16  32*256                =     4,096
static const size_t OFF_LC   = 7034880;  // f32  32*256                =     8,192
static const size_t OFF_HS   = 7043072;  // f32  32*32*256             =   262,144
// total ~7.3M float-slots ~= 29.2 MB

__device__ __forceinline__ float sigmoidf(float x) { return 1.0f / (1.0f + expf(-x)); }

// ---------------- STFT magnitude ----------------
__global__ void stft_kernel(const float* __restrict__ x, float* __restrict__ mag) {
  int p = blockIdx.x;   // 0..1023
  int t = blockIdx.y;   // 0..32
  int n = threadIdx.x;  // 0..63
  __shared__ float v[NPERSEG];
  int orig = t * NPERSEG + n - NPERSEG / 2;   // reflect pad (no edge repeat)
  if (orig < 0) orig = -orig;
  if (orig >= PATCH) orig = 2 * PATCH - 2 - orig;
  float win = 0.5f * (1.0f - cosf(6.28318530717958647692f * (float)n / (float)NPERSEG));
  v[n] = x[(size_t)p * PATCH + orig] * win;
  __syncthreads();
  if (n < NFREQ) {
    float re = 0.f, im = 0.f;
    float w = 6.28318530717958647692f * (float)n / (float)NPERSEG;
    for (int j = 0; j < NPERSEG; ++j) {
      float s, c;
      __sincosf(w * (float)j, &s, &c);
      re += v[j] * c;
      im -= v[j] * s;
    }
    mag[((size_t)p * NFREQ + n) * NFRAME + t] = sqrtf(re * re + im * im);
  }
}

// ---------------- PCC graph + node-importance mask ----------------
__global__ void graph_kernel(const float* __restrict__ mag, const float* __restrict__ ni_w,
                             const float* __restrict__ ni_b, float* __restrict__ hi) {
  const int N = NFREQ;
  int p = blockIdx.x, tid = threadIdx.x;
  __shared__ float m_s[N * N];
  __shared__ float c_s[N * N];
  __shared__ float nrm[N];
  __shared__ float ni[N];
  for (int i = tid; i < N * N; i += blockDim.x) m_s[i] = mag[(size_t)p * N * N + i];
  __syncthreads();
  if (tid < N) {
    float mean = 0.f, niv = ni_b[0];
    for (int f = 0; f < N; ++f) { mean += m_s[tid * N + f]; niv += m_s[tid * N + f] * ni_w[f]; }
    mean *= (1.0f / (float)N);
    float ss = 0.f;
    for (int f = 0; f < N; ++f) { float cc = m_s[tid * N + f] - mean; c_s[tid * N + f] = cc; ss += cc * cc; }
    nrm[tid] = sqrtf(ss);
    ni[tid] = niv;
  }
  __syncthreads();
  if (tid < N) {
    float acc = 0.f;
    for (int m = 0; m < N; ++m) {
      float dot = 0.f;
      for (int f = 0; f < N; ++f) dot += c_s[tid * N + f] * c_s[m * N + f];
      acc += dot / (nrm[tid] * nrm[m]) * ni[m];
    }
    hi[(size_t)p * N + tid] = (acc > 0.f) ? 1.0f : 0.0f;
  }
}

// ---------------- GAT layer (one graph per workgroup, LDS-resident) ----------------
template <int FIN, typename OUT_T>
__global__ void gat_kernel(const float* __restrict__ hin, const float* __restrict__ hi,
                           const float* __restrict__ W, const float* __restrict__ bvec,
                           const float* __restrict__ avec, const float* __restrict__ ab,
                           OUT_T* __restrict__ hout) {
  const int N = NFREQ, FOUT = 64;
  int p = blockIdx.x, tid = threadIdx.x;
  __shared__ float h_s[N * FIN];
  __shared__ float Wh[N * FOUT];
  __shared__ float att[N * N];
  __shared__ float hi_s[N];
  __shared__ float s1[N], s2[N], mrow[N], den[N];

  for (int i = tid; i < N * FIN; i += blockDim.x) h_s[i] = hin[(size_t)p * N * FIN + i];
  if (tid < N) hi_s[tid] = hi[(size_t)p * N + tid];
  __syncthreads();

  for (int idx = tid; idx < N * FOUT; idx += blockDim.x) {
    int i = idx / FOUT, j = idx % FOUT;
    float acc = bvec[j];
    for (int k = 0; k < FIN; ++k) acc += h_s[i * FIN + k] * W[k * FOUT + j];
    Wh[idx] = acc;
  }
  __syncthreads();

  if (tid < N) {
    float a1 = 0.f, a2 = 0.f;
    for (int j = 0; j < FOUT; ++j) {
      a1 += Wh[tid * FOUT + j] * avec[j];
      a2 += Wh[tid * FOUT + j] * avec[FOUT + j];
    }
    s1[tid] = a1; s2[tid] = a2;
  }
  __syncthreads();

  float abv = ab[0];
  if (tid < N) {
    float m = -1e30f;
    for (int k = 0; k < N; ++k) {
      float e = s1[tid] + s2[k] + abv;
      e = (e > 0.f) ? e : LRELU_A * e;
      m = fmaxf(m, e);
    }
    float d = 0.f;
    for (int k = 0; k < N; ++k) {
      float e = s1[tid] + s2[k] + abv;
      e = (e > 0.f) ? e : LRELU_A * e;
      d += expf(e - m);
    }
    mrow[tid] = m; den[tid] = d;
  }
  __syncthreads();

  // softmax over all j, THEN adjacency mask (faithful to reference)
  for (int idx = tid; idx < N * N; idx += blockDim.x) {
    int i = idx / N, k = idx % N;
    float e = s1[i] + s2[k] + abv;
    e = (e > 0.f) ? e : LRELU_A * e;
    att[idx] = expf(e - mrow[i]) / den[i] * hi_s[i] * hi_s[k];
  }
  __syncthreads();

  for (int idx = tid; idx < N * FOUT; idx += blockDim.x) {
    int i = idx / FOUT, j = idx % FOUT;
    float acc = 0.f;
    for (int k = 0; k < N; ++k) acc += att[i * N + k] * Wh[k * FOUT + j];
    acc = (acc > 0.f) ? acc : (expf(acc) - 1.0f);   // ELU
    hout[(size_t)p * N * FOUT + idx] = (OUT_T)acc;
  }
}

// ---------------- weight preparation (once per launch, outside hot loops) ----------------
// (K,N) f32 -> (N,K) f16 transposed
__global__ void cvt_transpose_kernel(const float* __restrict__ W, _Float16* __restrict__ out,
                                     int K, int N) {
  int idx = blockIdx.x * blockDim.x + threadIdx.x;
  if (idx >= K * N) return;
  int n = idx / K, k = idx - n * K;
  out[idx] = (_Float16)W[(size_t)k * N + n];
}

__global__ void cvt_kernel(const float* __restrict__ W, _Float16* __restrict__ out, int n) {
  int i = blockIdx.x * blockDim.x + threadIdx.x;
  if (i < n) out[i] = (_Float16)W[i];
}

// ---------------- WMMA f16 fragment loaders (CDNA5 wave32 layouts, no cvt) ----------------
// A 16x32: lanes 0-15 row M=lane, K={0..7,16..23}; lanes 16-31 same rows, K={8..15,24..31}
__device__ __forceinline__ v16h load_a_frag(const _Float16* __restrict__ A, int rowStride,
                                            int row0, int k0, int lane) {
  const _Float16* ar = A + (size_t)(row0 + (lane & 15)) * rowStride + k0 + ((lane < 16) ? 0 : 8);
  v16h r;
#pragma unroll
  for (int i = 0; i < 8; ++i) r[i] = ar[i];          // K = kb .. kb+7
#pragma unroll
  for (int i = 0; i < 8; ++i) r[8 + i] = ar[16 + i]; // K = kb+16 .. kb+23
  return r;
}

// B 32x16 from f16 (N,K) transposed weights: lanes 0-15 col N=lane K={0..7,16..23}; lanes 16-31 K={8..15,24..31}
__device__ __forceinline__ v16h load_b_frag(const _Float16* __restrict__ Wt, int ldk,
                                            int k0, int col0, int lane) {
  const _Float16* wr = Wt + (size_t)(col0 + (lane & 15)) * ldk + k0 + ((lane < 16) ? 0 : 8);
  v16h r;
#pragma unroll
  for (int i = 0; i < 8; ++i) r[i] = wr[i];          // K = kb .. kb+7
#pragma unroll
  for (int i = 0; i < 8; ++i) r[8 + i] = wr[16 + i]; // K = kb+16 .. kb+23
  return r;
}

// ---------------- encoder GEMM: C = [relu](A @ Wt^T + b), one 16x16 tile per wave ----------------
template <bool RELU>
__global__ void gemm_bias_kernel(const _Float16* __restrict__ A, const _Float16* __restrict__ Wt,
                                 const float* __restrict__ bias, _Float16* __restrict__ C,
                                 int M, int K, int N) {
  int lane = threadIdx.x & 31;
  int wave = threadIdx.x >> 5;
  int row0 = (blockIdx.y * 8 + wave) * 16;
  int col0 = blockIdx.x * 16;
  if (row0 >= M) return;
  const _Float16* aPre = A + (size_t)(row0 + (lane & 15)) * K;
  const _Float16* bPre = Wt + (size_t)(col0 + (lane & 15)) * K;
  v8f acc = {};
  for (int k0 = 0; k0 < K; k0 += 32) {
    if (k0 + 32 < K) {                       // gfx1250 global_prefetch of next K tile
      __builtin_prefetch(aPre + k0 + 32, 0, 3);
      __builtin_prefetch(bPre + k0 + 32, 0, 3);
    }
    v16h a = load_a_frag(A, K, row0, k0, lane);
    v16h b = load_b_frag(Wt, K, k0, col0, lane);
    acc = __builtin_amdgcn_wmma_f32_16x16x32_f16(false, a, false, b, (short)0, acc, false, false);
  }
  int col = col0 + (lane & 15);
  float bv = bias[col];
  int rbase = row0 + ((lane < 16) ? 0 : 8);
#pragma unroll
  for (int r = 0; r < 8; ++r) {
    float vv = acc[r] + bv;
    if (RELU) vv = fmaxf(vv, 0.0f);
    C[(size_t)(rbase + r) * N + col] = (_Float16)vv;
  }
}

// ---------------- LSTM gate GEMM: G = X_t @ wih^T + h @ whh^T + bih + bhh ----------------
__global__ void lstm_gate_kernel(const _Float16* __restrict__ Xt,   // H3 + t*128, row stride 32*128
                                 const _Float16* __restrict__ hstate,
                                 const _Float16* __restrict__ wih, const _Float16* __restrict__ whh,
                                 const float* __restrict__ bih, const float* __restrict__ bhh,
                                 float* __restrict__ G) {
  int lane = threadIdx.x & 31;
  int wave = threadIdx.x >> 5;        // 0..1 -> batch rows 0-15 / 16-31
  int row0 = wave * 16;
  int col0 = blockIdx.x * 16;         // gate column tile (0..63)
  v8f acc = {};
#pragma unroll
  for (int k0 = 0; k0 < 128; k0 += 32) {
    v16h a = load_a_frag(Xt, NUM_PATCH * 128, row0, k0, lane);
    v16h b = load_b_frag(wih, 128, k0, col0, lane);
    acc = __builtin_amdgcn_wmma_f32_16x16x32_f16(false, a, false, b, (short)0, acc, false, false);
  }
#pragma unroll
  for (int k0 = 0; k0 < 256; k0 += 32) {
    v16h a = load_a_frag(hstate, 256, row0, k0, lane);
    v16h b = load_b_frag(whh, 256, k0, col0, lane);
    acc = __builtin_amdgcn_wmma_f32_16x16x32_f16(false, a, false, b, (short)0, acc, false, false);
  }
  int col = col0 + (lane & 15);
  float bv = bih[col] + bhh[col];
  int rbase = row0 + ((lane < 16) ? 0 : 8);
#pragma unroll
  for (int r = 0; r < 8; ++r)
    G[(size_t)(rbase + r) * 1024 + col] = acc[r] + bv;
}

__global__ void lstm_point_kernel(const float* __restrict__ G, _Float16* __restrict__ h,
                                  float* __restrict__ c, float* __restrict__ hs, int t) {
  int idx = blockIdx.x * blockDim.x + threadIdx.x;
  if (idx >= 32 * 256) return;
  int b = idx >> 8, u = idx & 255;
  const float* g = G + (size_t)b * 1024;
  float gi = g[u], gf = g[256 + u], gg = g[512 + u], go = g[768 + u];
  float cc = sigmoidf(gf) * c[idx] + sigmoidf(gi) * tanhf(gg);
  float hh = sigmoidf(go) * tanhf(cc);
  c[idx] = cc;
  h[idx] = (_Float16)hh;
  hs[(size_t)(b * NUM_PATCH + t) * 256 + u] = hh;
}

__global__ void zero_f32_kernel(float* __restrict__ p, int n) {
  int i = blockIdx.x * blockDim.x + threadIdx.x;
  if (i < n) p[i] = 0.0f;
}
__global__ void zero_f16_kernel(_Float16* __restrict__ p, int n) {
  int i = blockIdx.x * blockDim.x + threadIdx.x;
  if (i < n) p[i] = (_Float16)0.0f;
}

__global__ void final_kernel(const float* __restrict__ hs, const float* __restrict__ out_w,
                             const float* __restrict__ out_b, float* __restrict__ out) {
  int b = blockIdx.x, tid = threadIdx.x;
  __shared__ float red[256];
  float acc = 0.f;
  for (int i = tid; i < NUM_PATCH * 256; i += 256)
    acc += hs[(size_t)b * NUM_PATCH * 256 + i] * out_w[i];
  red[tid] = acc;
  __syncthreads();
  for (int s = 128; s > 0; s >>= 1) {
    if (tid < s) red[tid] += red[tid + s];
    __syncthreads();
  }
  if (tid == 0) out[b] = red[0] + out_b[0];
}

extern "C" void kernel_launch(void* const* d_in, const int* in_sizes, int n_in,
                              void* d_out, int out_size, void* d_ws, size_t ws_size,
                              hipStream_t stream) {
  (void)in_sizes; (void)n_in; (void)out_size; (void)ws_size;
  const float* x        = (const float*)d_in[0];
  const float* gat_w0   = (const float*)d_in[1];
  const float* gat_b0   = (const float*)d_in[2];
  const float* gat_a0   = (const float*)d_in[3];
  const float* gat_ab0  = (const float*)d_in[4];
  const float* gat_w1   = (const float*)d_in[5];
  const float* gat_b1   = (const float*)d_in[6];
  const float* gat_a1   = (const float*)d_in[7];
  const float* gat_ab1  = (const float*)d_in[8];
  const float* ni_w     = (const float*)d_in[9];
  const float* ni_b     = (const float*)d_in[10];
  const float* enc_w0   = (const float*)d_in[11];
  const float* enc_b0   = (const float*)d_in[12];
  const float* enc_w1   = (const float*)d_in[13];
  const float* enc_b1   = (const float*)d_in[14];
  const float* enc_w2   = (const float*)d_in[15];
  const float* enc_b2   = (const float*)d_in[16];
  const float* enc_w3   = (const float*)d_in[17];
  const float* enc_b3   = (const float*)d_in[18];
  const float* lstm_wih = (const float*)d_in[19];
  const float* lstm_whh = (const float*)d_in[20];
  const float* lstm_bih = (const float*)d_in[21];
  const float* lstm_bhh = (const float*)d_in[22];
  const float* out_w    = (const float*)d_in[23];
  const float* out_b    = (const float*)d_in[24];
  float* out = (float*)d_out;
  float* ws  = (float*)d_ws;

  float*      mag  = ws + OFF_MAG;
  float*      hi   = ws + OFF_HI;
  float*      h1   = ws + OFF_H1;
  _Float16*   h2   = (_Float16*)(ws + OFF_H2);    // Y: (1024, 2112) f16
  _Float16*   w0t  = (_Float16*)(ws + OFF_W0T);   // (1024, 2112)
  _Float16*   w1t  = (_Float16*)(ws + OFF_W1T);   // (512, 1024)
  _Float16*   w2t  = (_Float16*)(ws + OFF_W2T);   // (256, 512)
  _Float16*   w3t  = (_Float16*)(ws + OFF_W3T);   // (128, 256)
  _Float16*   wihh = (_Float16*)(ws + OFF_WIH);   // (1024, 128)
  _Float16*   whhh = (_Float16*)(ws + OFF_WHH);   // (1024, 256)
  _Float16*   H0   = (_Float16*)(ws + OFF_H0);
  _Float16*   H1   = (_Float16*)(ws + OFF_HE1);
  _Float16*   H2   = (_Float16*)(ws + OFF_HE2);
  _Float16*   H3   = (_Float16*)(ws + OFF_HE3);
  float*      G    = ws + OFF_G;
  _Float16*   lh   = (_Float16*)(ws + OFF_LHH);
  float*      lc   = ws + OFF_LC;
  float*      hs   = ws + OFF_HS;

  // 0) weight preparation: f32 -> f16, transposed to (N,K) for contiguous B fragments
  cvt_transpose_kernel<<<(2112 * 1024 + 255) / 256, 256, 0, stream>>>(enc_w0, w0t, 2112, 1024);
  cvt_transpose_kernel<<<(1024 * 512 + 255) / 256, 256, 0, stream>>>(enc_w1, w1t, 1024, 512);
  cvt_transpose_kernel<<<(512 * 256 + 255) / 256, 256, 0, stream>>>(enc_w2, w2t, 512, 256);
  cvt_transpose_kernel<<<(256 * 128 + 255) / 256, 256, 0, stream>>>(enc_w3, w3t, 256, 128);
  cvt_kernel<<<(1024 * 128 + 255) / 256, 256, 0, stream>>>(lstm_wih, wihh, 1024 * 128);
  cvt_kernel<<<(1024 * 256 + 255) / 256, 256, 0, stream>>>(lstm_whh, whhh, 1024 * 256);

  // 1) STFT magnitudes
  stft_kernel<<<dim3(NPATCHES, NFRAME), NPERSEG, 0, stream>>>(x, mag);
  // 2) PCC graph + adjacency mask vector hi
  graph_kernel<<<NPATCHES, 256, 0, stream>>>(mag, ni_w, ni_b, hi);
  // 3) two GAT layers (ELU inside); GAT-1 emits f16 Y directly
  gat_kernel<NFREQ, float><<<NPATCHES, 256, 0, stream>>>(mag, hi, gat_w0, gat_b0, gat_a0, gat_ab0, h1);
  gat_kernel<64, _Float16><<<NPATCHES, 256, 0, stream>>>(h1, hi, gat_w1, gat_b1, gat_a1, gat_ab1, h2);
  // 4) encoder MLP via WMMA, f16 activations end-to-end, f32 accumulate
  gemm_bias_kernel<true ><<<dim3(1024 / 16, 1024 / 128), 256, 0, stream>>>(h2, w0t, enc_b0, H0, 1024, 2112, 1024);
  gemm_bias_kernel<true ><<<dim3(512 / 16, 1024 / 128), 256, 0, stream>>>(H0, w1t, enc_b1, H1, 1024, 1024, 512);
  gemm_bias_kernel<true ><<<dim3(256 / 16, 1024 / 128), 256, 0, stream>>>(H1, w2t, enc_b2, H2, 1024, 512, 256);
  gemm_bias_kernel<false><<<dim3(128 / 16, 1024 / 128), 256, 0, stream>>>(H2, w3t, enc_b3, H3, 1024, 256, 128);
  // 5) LSTM over 32 timesteps
  zero_f16_kernel<<<(32 * 256 + 255) / 256, 256, 0, stream>>>(lh, 32 * 256);
  zero_f32_kernel<<<(32 * 256 + 255) / 256, 256, 0, stream>>>(lc, 32 * 256);
  for (int t = 0; t < NUM_PATCH; ++t) {
    lstm_gate_kernel<<<64, 64, 0, stream>>>(H3 + (size_t)t * 128, lh,
                                            wihh, whhh, lstm_bih, lstm_bhh, G);
    lstm_point_kernel<<<32, 256, 0, stream>>>(G, lh, lc, hs, t);
  }
  // 6) output head
  final_kernel<<<32, 256, 0, stream>>>(hs, out_w, out_b, out);
}
